// UpdateLayer2UU_29145648071071
// MI455X (gfx1250) — matrix operand order repaired and checked by reading, not compiled, not run on hardware
//
#include <hip/hip_runtime.h>
#include <hip/hip_bf16.h>

typedef _Float16 f16;
typedef __attribute__((ext_vector_type(16))) _Float16 v16h;
typedef __attribute__((ext_vector_type(8)))  _Float16 v8h;
typedef __attribute__((ext_vector_type(8)))  float    v8f;

// ---------------------------------------------------------------------------
// CDNA5 async global->LDS copy (ASYNCcnt path, bypasses VGPRs).
// Guarded: falls back to plain element copy if the builtin is absent.
// Probe feedback: builtin signature is (v4i* src, v4i* dst, imm off, imm cpol)
// with v4i = int __attribute__((vector_size(16))).
// ---------------------------------------------------------------------------
#if defined(__has_builtin)
#if __has_builtin(__builtin_amdgcn_global_load_async_to_lds_b128)
#define HAVE_ASYNC_LDS 1
#endif
#endif

typedef __attribute__((__vector_size__(16))) int v4i_vs;

#ifdef HAVE_ASYNC_LDS
#define ASYNC_COPY_B128(srcp, dstp)                                  \
  __builtin_amdgcn_global_load_async_to_lds_b128(                    \
      (v4i_vs*)(uintptr_t)(srcp), (v4i_vs*)(uintptr_t)(dstp), 0, 0)
#endif

__device__ __forceinline__ void wait_async0() {
#if defined(__has_builtin) && __has_builtin(__builtin_amdgcn_s_wait_asynccnt)
  __builtin_amdgcn_s_wait_asynccnt(0);
#else
  asm volatile("s_wait_asynccnt 0x0" ::: "memory");
#endif
}

// ---------------------------------------------------------------------------
// WMMA helpers (CDNA5 v_wmma_f32_16x16x32_f16, wave32)
// ---------------------------------------------------------------------------
__device__ __forceinline__ v8f wmma32(v16h a, v16h b, v8f c) {
  // 8 args: (neg_a, A, neg_b, B, c_mod, C, reuse_a, reuse_b)
  return __builtin_amdgcn_wmma_f32_16x16x32_f16(false, a, false, b, (short)0, c,
                                                false, false);
}

// A-matrix 16x32 f16 loader from row-major f16 buffer:
// lane m = lane&15, g = lane>>4; VGPR i(0..3): K = g*8 + 2i{,+1};
// VGPR i(4..7): K = 16 + g*8 + 2(i-4){,+1}.   (ISA 7.12.2)
__device__ __forceinline__ v16h loadA16(const f16* p, int ld, int m0, int k0,
                                        int lane) {
  const int m = lane & 15, g = lane >> 4;
  const f16* row = p + (m0 + m) * ld + k0;
  v16h a;
#pragma unroll
  for (int i = 0; i < 4; ++i) {
    int k = g * 8 + 2 * i;
    a[2 * i] = row[k];
    a[2 * i + 1] = row[k + 1];
  }
#pragma unroll
  for (int i = 0; i < 4; ++i) {
    int k = 16 + g * 8 + 2 * i;
    a[8 + 2 * i] = row[k];
    a[9 + 2 * i] = row[k + 1];
  }
  return a;
}

// B-matrix 32x16 loader where B is stored TRANSPOSED row-major (Bt[n][k]):
// lane half g holds K=g*8..g*8+7 then K=16+g*8..16+g*8+7 of column n.
__device__ __forceinline__ v16h loadBt16(const f16* p, int ld, int n0, int k0,
                                         int lane) {
  const int n = lane & 15, half = lane >> 4;
  const f16* row = p + (n0 + n) * ld + k0 + half * 8;
  v16h b;
#pragma unroll
  for (int j = 0; j < 16; ++j) b[j] = row[j < 8 ? j : j + 8];
  return b;
}

// ---------------------------------------------------------------------------
// rank-1 vectors: kc = ciW[256,:] @ Wk_du ; vc = ciW[256,:] @ Wv_du
// ---------------------------------------------------------------------------
__global__ void rank1_kernel(const float* __restrict__ ciW,
                             const float* __restrict__ Wk,
                             const float* __restrict__ Wv,
                             float* __restrict__ kc, float* __restrict__ vc) {
  const int j = threadIdx.x;
  const float* wl = ciW + 256 * 256;  // last row of [257,256]
  const float* W = (blockIdx.x == 0) ? Wk : Wv;
  float* o = (blockIdx.x == 0) ? kc : vc;
  float acc = 0.f;
#pragma unroll 8
  for (int i = 0; i < 256; ++i) acc += wl[i] * W[i * 256 + j];
  o[j] = acc;
}

// ---------------------------------------------------------------------------
// Generic WMMA GEMM:  out[r, col0+n] = f16( X[r,:] @ W[:, nb..] + bias [+res] )
// Block = 256 threads (8 waves, 4Mx2N), tile 64x64, K-chunks of 64.
// f16 inputs stream through GLOBAL_LOAD_ASYNC_TO_LDS_B128 when available.
// ---------------------------------------------------------------------------
template <typename TIN>
__global__ __launch_bounds__(256) void gemm_kernel(
    const TIN* __restrict__ X, int ldx, const float* __restrict__ W, int ldw,
    const float* __restrict__ bias, const float* __restrict__ residual,
    f16* __restrict__ out, int ldout, int col0, int K) {
  const int tid = threadIdx.x, lane = tid & 31, w = tid >> 5;
  const int wm = w >> 1, wn = w & 1;
  const int n16 = lane & 15, half = lane >> 4;
  const int rb = blockIdx.x * 64, nb = blockIdx.y * 64;

  __shared__ f16 sX[64][72];   // row stride 144B (16B aligned, async-friendly)
  __shared__ f16 sWt[64][72];

  v8f a0 = {}, a1 = {};
  const int nchunks = K >> 6;
  for (int kb = 0; kb < nchunks; ++kb) {
    __syncthreads();
    if (kb + 1 < nchunks) {
      // warm the caches for the next K-chunk (global_prefetch_b8)
      __builtin_prefetch(&W[(size_t)((kb + 1) * 64 + (tid >> 2)) * ldw + nb +
                            (tid & 3) * 16], 0, 1);
      if (tid < 64)
        __builtin_prefetch(&X[(size_t)(rb + tid) * ldx + (kb + 1) * 64], 0, 1);
    }
    if constexpr (sizeof(TIN) == 2) {
#ifdef HAVE_ASYNC_LDS
      // 64 rows x 128B tile = 512 x 16B async transfers (2 per thread)
#pragma unroll
      for (int e = tid; e < 512; e += 256) {
        const int r = e >> 3, c8 = (e & 7) * 8;
        const f16* src = (const f16*)X + (size_t)(rb + r) * ldx + kb * 64 + c8;
        ASYNC_COPY_B128(src, &sX[r][c8]);
      }
#else
#pragma unroll 4
      for (int e = tid; e < 4096; e += 256) {
        int r = e >> 6, c = e & 63;
        sX[r][c] = (f16)X[(size_t)(rb + r) * ldx + kb * 64 + c];
      }
#endif
    } else {
#pragma unroll 4
      for (int e = tid; e < 4096; e += 256) {
        int r = e >> 6, c = e & 63;
        sX[r][c] = (f16)X[(size_t)(rb + r) * ldx + kb * 64 + c];
      }
    }
#pragma unroll 4
    for (int e = tid; e < 4096; e += 256) {
      int k = e >> 6, c = e & 63;
      sWt[c][k] = (f16)W[(size_t)(kb * 64 + k) * ldw + nb + c];
    }
    if constexpr (sizeof(TIN) == 2) {
#ifdef HAVE_ASYNC_LDS
      wait_async0();
#endif
    }
    __syncthreads();
#pragma unroll
    for (int ks = 0; ks < 64; ks += 32) {
      v16h a = loadA16(&sX[0][0], 72, 16 * wm, ks, lane);
      v16h b0 = loadBt16(&sWt[0][0], 72, 32 * wn, ks, lane);
      v16h b1 = loadBt16(&sWt[0][0], 72, 32 * wn + 16, ks, lane);
      a0 = wmma32(a, b0, a0);
      a1 = wmma32(a, b1, a1);
    }
  }
  // epilogue (C/D layout: vgpr r -> row r + half*8, lane&15 -> n)
#pragma unroll
  for (int r = 0; r < 8; ++r) {
    const int m = rb + 16 * wm + r + half * 8;
    {
      const int n = nb + 32 * wn + n16;
      float v = a0[r] + (bias ? bias[n] : 0.f);
      if (residual) v += residual[(size_t)m * 256 + n];
      out[(size_t)m * ldout + col0 + n] = (f16)v;
    }
    {
      const int n = nb + 32 * wn + 16 + n16;
      float v = a1[r] + (bias ? bias[n] : 0.f);
      if (residual) v += residual[(size_t)m * 256 + n];
      out[(size_t)m * ldout + col0 + n] = (f16)v;
    }
  }
}

// ---------------------------------------------------------------------------
// Attention per (b,h): Q[64,32] x K[NK,32]^T -> softmax -> @ V[NK,32].
// Optional rank-1 CI correction (DU branch):
//   scores = (Q.Kb + CI*(Q.kc)) * scale ;  ctx += (sum_n P*CI) * vc
// 128 threads = 4 waves, wave w owns query rows 16w..16w+15.
// ---------------------------------------------------------------------------
template <int NK, bool USECI>
__global__ __launch_bounds__(128) void attn_kernel(
    const f16* __restrict__ Q, const f16* __restrict__ Kb,
    const f16* __restrict__ Vb, const float* __restrict__ CI,
    const float* __restrict__ kc, const float* __restrict__ vc,
    f16* __restrict__ ctx) {
  const int h = blockIdx.x, b = blockIdx.y;
  const int tid = threadIdx.x, lane = tid & 31, w = tid >> 5;
  const int n16 = lane & 15, half = lane >> 4;
  constexpr int NT = NK / 16;
  constexpr float scale = 0.17677669529663687f;  // 1/sqrt(32)

  __shared__ f16 sP[64][NK + 8];
  __shared__ f16 sVt[32][NK + 8];

  // stage V transposed: sVt[n][k] = V[k][n]
  for (int e = tid; e < NK * 32; e += 128) {
    int k = e >> 5, n = e & 31;
    sVt[n][k] = Vb[(size_t)(b * NK + k) * 256 + h * 32 + n];
  }
  __syncthreads();

  // Q A-operand straight from global (two 16B loads per lane)
  const f16* qrow = Q + (size_t)(b * 64 + 16 * w + n16) * 256 + h * 32;
  v8h qa0 = *(const v8h*)(qrow + half * 8);
  v8h qa1 = *(const v8h*)(qrow + 16 + half * 8);
  v16h a;
#pragma unroll
  for (int j = 0; j < 8; ++j) {
    a[j] = qa0[j];
    a[8 + j] = qa1[j];
  }

  float qdot = 0.f;
  if constexpr (USECI) {
#pragma unroll
    for (int j = 0; j < 8; ++j) {
      qdot += (float)qa0[j] * kc[h * 32 + half * 8 + j];
      qdot += (float)qa1[j] * kc[h * 32 + 16 + half * 8 + j];
    }
    qdot += __shfl_xor(qdot, 16, 32);  // combine K-halves of same row
  }

  // scores: one WMMA per 16-key tile (hd == one K step)
  v8f accs[NT];
#pragma unroll
  for (int nt = 0; nt < NT; ++nt) {
    const f16* krow =
        Kb + (size_t)(b * NK + nt * 16 + n16) * 256 + h * 32 + half * 8;
    v8h kb0 = *(const v8h*)krow;
    v8h kb1 = *(const v8h*)(krow + 16);
    v16h bv;
#pragma unroll
    for (int j = 0; j < 8; ++j) {
      bv[j] = kb0[j];
      bv[8 + j] = kb1[j];
    }
    v8f z = {};
    accs[nt] = wmma32(a, bv, z);
  }

  // scale (+ CI rank-1 correction)
#pragma unroll
  for (int nt = 0; nt < NT; ++nt) {
#pragma unroll
    for (int r = 0; r < 8; ++r) {
      float s = accs[nt][r];
      if constexpr (USECI) {
        float ci =
            CI[(size_t)(b * 64 + 16 * w + r + half * 8) * 64 + nt * 16 + n16];
        s += ci * qdot;
      }
      accs[nt][r] = s * scale;
    }
  }

  // register softmax: reductions across the 16 lanes of each half
  float tci[8];
#pragma unroll
  for (int r = 0; r < 8; ++r) {
    float m = -3.0e38f;
#pragma unroll
    for (int nt = 0; nt < NT; ++nt) m = fmaxf(m, accs[nt][r]);
#pragma unroll
    for (int msk = 8; msk >= 1; msk >>= 1) m = fmaxf(m, __shfl_xor(m, msk, 16));
    float s = 0.f;
#pragma unroll
    for (int nt = 0; nt < NT; ++nt) {
      float e = __expf(accs[nt][r] - m);
      accs[nt][r] = e;
      s += e;
    }
#pragma unroll
    for (int msk = 8; msk >= 1; msk >>= 1) s += __shfl_xor(s, msk, 16);
    const float inv = 1.f / s;
    float t = 0.f;
#pragma unroll
    for (int nt = 0; nt < NT; ++nt) {
      float p = accs[nt][r] * inv;
      if constexpr (USECI) {
        float ci =
            CI[(size_t)(b * 64 + 16 * w + r + half * 8) * 64 + nt * 16 + n16];
        t += p * ci;
      }
      sP[16 * w + r + half * 8][nt * 16 + n16] = (f16)p;
    }
    if constexpr (USECI) {
#pragma unroll
      for (int msk = 8; msk >= 1; msk >>= 1) t += __shfl_xor(t, msk, 16);
    }
    tci[r] = t;
  }
  __syncthreads();

  // ctx = P @ V (+ tci * vc)
#pragma unroll
  for (int ntv = 0; ntv < 2; ++ntv) {
    v8f acc = {};
#pragma unroll
    for (int kb = 0; kb < NK / 32; ++kb) {
      v16h ap = loadA16(&sP[0][0], NK + 8, 16 * w, kb * 32, lane);
      v16h bv = loadBt16(&sVt[0][0], NK + 8, ntv * 16, kb * 32, lane);
      acc = wmma32(ap, bv, acc);
    }
#pragma unroll
    for (int r = 0; r < 8; ++r) {
      const int m = 16 * w + r + half * 8;
      const int n = ntv * 16 + n16;
      float v = acc[r];
      if constexpr (USECI) v += tci[r] * vc[h * 32 + n];
      ctx[(size_t)(b * 64 + m) * 256 + h * 32 + n] = (f16)v;
    }
  }
}

// ---------------------------------------------------------------------------
// post: out = LayerNorm( relu( h1 @ post_W + post_b ) ) * g + beta   (fp32)
// Block = 256 threads, 64 rows x 256 cols; waves 4Mx2N (16 rows x 128 cols).
// ---------------------------------------------------------------------------
__global__ __launch_bounds__(256) void post_kernel(
    const f16* __restrict__ X, const float* __restrict__ W,
    const float* __restrict__ bias, const float* __restrict__ g,
    const float* __restrict__ beta, float* __restrict__ out) {
  const int tid = threadIdx.x, lane = tid & 31, w = tid >> 5;
  const int wm = w >> 1, wn = w & 1;
  const int n16 = lane & 15, half = lane >> 4;
  const int row0 = blockIdx.x * 64;

  __shared__ f16 sWt[256][40];
  __shared__ float pS[64][2], pQ[64][2];

  v8f acc[8];
#pragma unroll
  for (int t = 0; t < 8; ++t) acc[t] = v8f{};

  for (int kb = 0; kb < 8; ++kb) {  // K = 256 in chunks of 32
    __syncthreads();
#pragma unroll 4
    for (int e = tid; e < 256 * 32; e += 256) {
      int k = e >> 8, n = e & 255;
      sWt[n][k] = (f16)W[(size_t)(kb * 32 + k) * 256 + n];
    }
    __syncthreads();
    const f16* xrow = X + (size_t)(row0 + 16 * wm + n16) * 256 + kb * 32;
    v8h x0 = *(const v8h*)(xrow + half * 8);
    v8h x1 = *(const v8h*)(xrow + 16 + half * 8);
    v16h aop;
#pragma unroll
    for (int j = 0; j < 8; ++j) {
      aop[j] = x0[j];
      aop[8 + j] = x1[j];
    }
#pragma unroll
    for (int t = 0; t < 8; ++t) {
      v16h bv = loadBt16(&sWt[0][0], 40, wn * 128 + t * 16, 0, lane);
      acc[t] = wmma32(aop, bv, acc[t]);
    }
  }

  // bias + relu, per-row partial sums across this wave's 128 columns
#pragma unroll
  for (int r = 0; r < 8; ++r) {
    float s = 0.f, q = 0.f;
#pragma unroll
    for (int t = 0; t < 8; ++t) {
      const int n = wn * 128 + t * 16 + n16;
      float v = fmaxf(acc[t][r] + bias[n], 0.f);
      acc[t][r] = v;
      s += v;
      q += v * v;
    }
#pragma unroll
    for (int msk = 8; msk >= 1; msk >>= 1) {
      s += __shfl_xor(s, msk, 16);
      q += __shfl_xor(q, msk, 16);
    }
    if (n16 == 0) {
      const int m = 16 * wm + r + half * 8;
      pS[m][wn] = s;
      pQ[m][wn] = q;
    }
  }
  __syncthreads();

#pragma unroll
  for (int r = 0; r < 8; ++r) {
    const int m = 16 * wm + r + half * 8;
    const float s = pS[m][0] + pS[m][1];
    const float q = pQ[m][0] + pQ[m][1];
    const float mu = s * (1.f / 256.f);
    const float var = q * (1.f / 256.f) - mu * mu;
    const float inv = rsqrtf(var + 1e-5f);
#pragma unroll
    for (int t = 0; t < 8; ++t) {
      const int n = wn * 128 + t * 16 + n16;
      out[(size_t)(row0 + m) * 256 + n] =
          (acc[t][r] - mu) * inv * g[n] + beta[n];
    }
  }
}

// ---------------------------------------------------------------------------
// Host orchestration
// ---------------------------------------------------------------------------
extern "C" void kernel_launch(void* const* d_in, const int* in_sizes, int n_in,
                              void* d_out, int out_size, void* d_ws,
                              size_t ws_size, hipStream_t stream) {
  (void)in_sizes; (void)n_in; (void)out_size; (void)ws_size;
  // setup_inputs() dict order:
  const float* UU = (const float*)d_in[0];   // [32,64,256]
  const float* DU = (const float*)d_in[1];   // [32,64,256]
  const float* IN = (const float*)d_in[2];   // [32,128,256]
  const float* TA = (const float*)d_in[3];   // [32,128,256]
  const float* CI = (const float*)d_in[4];   // [32,64,64]
  // params: ta(5..12) uu(13..20) du(21..28) in(29..36)
  //   each mha: Wq,bq,Wk,bk,Wv,bv,Wo,bo
  const float* ta_Wq = (const float*)d_in[5];  const float* ta_bq = (const float*)d_in[6];
  const float* ta_Wk = (const float*)d_in[7];  const float* ta_bk = (const float*)d_in[8];
  const float* ta_Wv = (const float*)d_in[9];  const float* ta_bv = (const float*)d_in[10];
  const float* ta_Wo = (const float*)d_in[11]; const float* ta_bo = (const float*)d_in[12];
  const float* uu_Wq = (const float*)d_in[13]; const float* uu_bq = (const float*)d_in[14];
  const float* uu_Wk = (const float*)d_in[15]; const float* uu_bk = (const float*)d_in[16];
  const float* uu_Wv = (const float*)d_in[17]; const float* uu_bv = (const float*)d_in[18];
  const float* uu_Wo = (const float*)d_in[19]; const float* uu_bo = (const float*)d_in[20];
  const float* du_Wq = (const float*)d_in[21]; const float* du_bq = (const float*)d_in[22];
  const float* du_Wk = (const float*)d_in[23]; const float* du_bk = (const float*)d_in[24];
  const float* du_Wv = (const float*)d_in[25]; const float* du_bv = (const float*)d_in[26];
  const float* du_Wo = (const float*)d_in[27]; const float* du_bo = (const float*)d_in[28];
  const float* in_Wq = (const float*)d_in[29]; const float* in_bq = (const float*)d_in[30];
  const float* in_Wk = (const float*)d_in[31]; const float* in_bk = (const float*)d_in[32];
  const float* in_Wv = (const float*)d_in[33]; const float* in_bv = (const float*)d_in[34];
  const float* in_Wo = (const float*)d_in[35]; const float* in_bo = (const float*)d_in[36];
  const float* ciW    = (const float*)d_in[37];  // [257,256]
  const float* cib    = (const float*)d_in[38];
  const float* fuseW  = (const float*)d_in[39];  // [1024,256]
  const float* fuseb  = (const float*)d_in[40];
  const float* postW  = (const float*)d_in[41];
  const float* postb  = (const float*)d_in[42];
  const float* lng    = (const float*)d_in[43];
  const float* lnb    = (const float*)d_in[44];
  float* OUT = (float*)d_out;

  const int RUU = 2048, RKV = 4096;  // B*64, B*128
  char* ws = (char*)d_ws;
  auto alloc = [&](size_t bytes) -> void* {
    void* p = ws;
    ws += (bytes + 255) & ~(size_t)255;
    return p;
  };
  f16* q_ta = (f16*)alloc((size_t)RUU * 256 * 2);
  f16* q_uu = (f16*)alloc((size_t)RUU * 256 * 2);
  f16* q_in = (f16*)alloc((size_t)RUU * 256 * 2);
  f16* q_du = (f16*)alloc((size_t)RUU * 256 * 2);
  f16* k_ta = (f16*)alloc((size_t)RKV * 256 * 2);
  f16* v_ta = (f16*)alloc((size_t)RKV * 256 * 2);
  f16* k_uu = (f16*)alloc((size_t)RUU * 256 * 2);
  f16* v_uu = (f16*)alloc((size_t)RUU * 256 * 2);
  f16* k_in = (f16*)alloc((size_t)RKV * 256 * 2);
  f16* v_in = (f16*)alloc((size_t)RKV * 256 * 2);
  f16* dufx = (f16*)alloc((size_t)RUU * 256 * 2);
  f16* k_du = (f16*)alloc((size_t)RUU * 256 * 2);
  f16* v_du = (f16*)alloc((size_t)RUU * 256 * 2);
  f16* c_du = (f16*)alloc((size_t)RUU * 256 * 2);
  f16* c_ta = (f16*)alloc((size_t)RUU * 256 * 2);
  f16* c_uu = (f16*)alloc((size_t)RUU * 256 * 2);
  f16* c_in = (f16*)alloc((size_t)RUU * 256 * 2);
  f16* cat  = (f16*)alloc((size_t)RUU * 1024 * 2);
  f16* h1   = (f16*)alloc((size_t)RUU * 256 * 2);
  float* kc = (float*)alloc(256 * 4);
  float* vc = (float*)alloc(256 * 4);

  const dim3 blk(256);
  const dim3 gUU(RUU / 64, 4), gKV(RKV / 64, 4);

  rank1_kernel<<<2, 256, 0, stream>>>(ciW, du_Wk, du_Wv, kc, vc);

  // Q projections (all from UUMat)
  gemm_kernel<float><<<gUU, blk, 0, stream>>>(UU, 256, ta_Wq, 256, ta_bq, nullptr, q_ta, 256, 0, 256);
  gemm_kernel<float><<<gUU, blk, 0, stream>>>(UU, 256, uu_Wq, 256, uu_bq, nullptr, q_uu, 256, 0, 256);
  gemm_kernel<float><<<gUU, blk, 0, stream>>>(UU, 256, in_Wq, 256, in_bq, nullptr, q_in, 256, 0, 256);
  gemm_kernel<float><<<gUU, blk, 0, stream>>>(UU, 256, du_Wq, 256, du_bq, nullptr, q_du, 256, 0, 256);
  // K/V projections
  gemm_kernel<float><<<gKV, blk, 0, stream>>>(TA, 256, ta_Wk, 256, ta_bk, nullptr, k_ta, 256, 0, 256);
  gemm_kernel<float><<<gKV, blk, 0, stream>>>(TA, 256, ta_Wv, 256, ta_bv, nullptr, v_ta, 256, 0, 256);
  gemm_kernel<float><<<gUU, blk, 0, stream>>>(UU, 256, uu_Wk, 256, uu_bk, nullptr, k_uu, 256, 0, 256);
  gemm_kernel<float><<<gUU, blk, 0, stream>>>(UU, 256, uu_Wv, 256, uu_bv, nullptr, v_uu, 256, 0, 256);
  gemm_kernel<float><<<gKV, blk, 0, stream>>>(IN, 256, in_Wk, 256, in_bk, nullptr, k_in, 256, 0, 256);
  gemm_kernel<float><<<gKV, blk, 0, stream>>>(IN, 256, in_Wv, 256, in_bv, nullptr, v_in, 256, 0, 256);
  // DU branch: du_fixed = DU @ ciW[:256] + cib ; K/V base from du_fixed
  gemm_kernel<float><<<gUU, blk, 0, stream>>>(DU, 256, ciW, 256, cib, nullptr, dufx, 256, 0, 256);
  gemm_kernel<f16><<<gUU, blk, 0, stream>>>(dufx, 256, du_Wk, 256, du_bk, nullptr, k_du, 256, 0, 256);
  gemm_kernel<f16><<<gUU, blk, 0, stream>>>(dufx, 256, du_Wv, 256, du_bv, nullptr, v_du, 256, 0, 256);

  // attention (grid = H x B)
  const dim3 gA(8, 32), blkA(128);
  attn_kernel<64, true><<<gA, blkA, 0, stream>>>(q_du, k_du, v_du, CI, kc, vc, c_du);
  attn_kernel<128, false><<<gA, blkA, 0, stream>>>(q_ta, k_ta, v_ta, nullptr, nullptr, nullptr, c_ta);
  attn_kernel<64, false><<<gA, blkA, 0, stream>>>(q_uu, k_uu, v_uu, nullptr, nullptr, nullptr, c_uu);
  attn_kernel<128, false><<<gA, blkA, 0, stream>>>(q_in, k_in, v_in, nullptr, nullptr, nullptr, c_in);

  // output projections written directly into cat layout [2048,1024]
  gemm_kernel<f16><<<gUU, blk, 0, stream>>>(c_du, 256, du_Wo, 256, du_bo, nullptr, cat, 1024, 0, 256);
  gemm_kernel<f16><<<gUU, blk, 0, stream>>>(c_ta, 256, ta_Wo, 256, ta_bo, nullptr, cat, 1024, 256, 256);
  gemm_kernel<f16><<<gUU, blk, 0, stream>>>(c_uu, 256, uu_Wo, 256, uu_bo, nullptr, cat, 1024, 512, 256);
  gemm_kernel<f16><<<gUU, blk, 0, stream>>>(c_in, 256, in_Wo, 256, in_bo, nullptr, cat, 1024, 768, 256);

  // fuse (K=1024) + residual(UU)
  gemm_kernel<f16><<<gUU, blk, 0, stream>>>(cat, 1024, fuseW, 256, fuseb, UU, h1, 256, 0, 1024);

  // post linear + relu + layernorm -> fp32 output
  post_kernel<<<RUU / 64, blk, 0, stream>>>(h1, postW, postb, lng, lnb, OUT);
}
